// WindowAttention_24077586661776
// MI455X (gfx1250) — compile-verified, matrix operand order checked by reading, not run on hardware
//
#include <hip/hip_runtime.h>
#include <hip/hip_bf16.h>

typedef __attribute__((ext_vector_type(16))) __bf16 v16bf;
typedef __attribute__((ext_vector_type(8)))  float  v8f;
typedef __attribute__((ext_vector_type(4)))  int    v4i;

#define DIMC   384
#define NHEADS 12
#define NTOK   343
#define NPAD   352
#define BWIN   128
#define ROWS   (BWIN * NTOK)      // 43904
#define NN     (NTOK * NTOK)      // 117649
#define TBL    2197
#define CPBH   512

#define AS1 __attribute__((address_space(1)))
#define AS3 __attribute__((address_space(3)))

#if defined(__HIP_DEVICE_COMPILE__) && __has_builtin(__builtin_amdgcn_global_load_async_to_lds_b128)
#define HAVE_ASYNC_LDS 1
#else
#define HAVE_ASYNC_LDS 0
#endif

// ---------- helpers ----------

__device__ inline __bf16 f2bf(float x) {
  unsigned u = __float_as_uint(x);
  unsigned r = (u + 0x7FFFu + ((u >> 16) & 1u)) >> 16;
  unsigned short s = (unsigned short)r;
  __bf16 b;
  __builtin_memcpy(&b, &s, 2);
  return b;
}

// async 16-byte global->LDS copy (CDNA5 GLOBAL_LOAD_ASYNC_TO_LDS_B128)
__device__ inline void async_cp16(const void* g, void* l) {
#if HAVE_ASYNC_LDS
  __builtin_amdgcn_global_load_async_to_lds_b128(
      (AS1 v4i*)(g), (AS3 v4i*)(l), 0, 0);
#else
  (void)g; (void)l;
#endif
}

__device__ inline void async_wait() {
#if HAVE_ASYNC_LDS
#if __has_builtin(__builtin_amdgcn_s_wait_asynccnt)
  __builtin_amdgcn_s_wait_asynccnt(0);
#else
  asm volatile("s_wait_asynccnt 0x0" ::: "memory");
#endif
#endif
}

// Load a 16x32 bf16 A/B fragment (row-major source, per ISA 7.12.2):
// lanes 0-15 hold row (lane), K = {0..7, 16..23}; lanes 16-31 hold row
// (lane-16), K = {8..15, 24..31}. base points at (row0, k0).
__device__ inline v16bf ld_frag(const __bf16* base, int stride) {
  const int lane = threadIdx.x & 31;
  const __bf16* p = base + (lane & 15) * stride + ((lane >> 4) << 3);
  v16bf f;
#pragma unroll
  for (int i = 0; i < 8; ++i) { f[i] = p[i]; f[i + 8] = p[16 + i]; }
  return f;
}

__device__ inline v8f wmma_bf16(v16bf a, v16bf b, v8f c) {
  return __builtin_amdgcn_wmma_f32_16x16x32_bf16(
      /*neg_a=*/false, a, /*neg_b=*/false, b,
      /*c_mod=*/(short)0, c, /*reuse_a=*/false, /*reuse_b=*/false);
}

// ---------- K1: weights fp32 -> bf16 ----------

__global__ void k_convert(const float* __restrict__ a, const float* __restrict__ b,
                          __bf16* __restrict__ oa, __bf16* __restrict__ ob) {
  int i = blockIdx.x * 256 + threadIdx.x;
  if (i < 3 * DIMC * DIMC) oa[i] = f2bf(a[i]);
  if (i < DIMC * DIMC)     ob[i] = f2bf(b[i]);
}

// ---------- K2: CPB-MLP  t[e][h] = relu(coord@w1^T + b1)@w2^T ----------

__global__ void k_cpb(const float* __restrict__ table, const float* __restrict__ w1,
                      const float* __restrict__ b1, const float* __restrict__ w2,
                      float* __restrict__ t) {
  int i = blockIdx.x * 256 + threadIdx.x;
  if (i >= TBL * NHEADS) return;
  int e = i / NHEADS, h = i % NHEADS;
  float c0 = table[e * 3 + 0], c1 = table[e * 3 + 1], c2 = table[e * 3 + 2];
  float acc = 0.f;
  for (int j = 0; j < CPBH; ++j) {
    float hv = fmaf(w1[j * 3 + 0], c0, fmaf(w1[j * 3 + 1], c1, fmaf(w1[j * 3 + 2], c2, b1[j])));
    hv = fmaxf(hv, 0.f);
    acc = fmaf(w2[h * CPBH + j], hv, acc);
  }
  t[i] = acc;
}

// ---------- K3: bias_full[h][m][n] = 16*sigmoid(t[rpi[m*N+n]][h]) ----------

__global__ void k_bias(const float* __restrict__ t, const int* __restrict__ rpi,
                       float* __restrict__ bias) {
  int i = blockIdx.x * 256 + threadIdx.x;
  if (i >= NHEADS * NN) return;
  int h = i / NN, r = i % NN;
  float x = t[rpi[r] * NHEADS + h];
  bias[i] = 16.f / (1.f + __expf(-x));
}

// ---------- K4: QKV GEMM + bias + per-head q/k normalization ----------

__global__ void __launch_bounds__(128)
k_qkv_gemm(const float* __restrict__ x, const __bf16* __restrict__ wq,
           const float* __restrict__ q_bias, const float* __restrict__ v_bias,
           __bf16* __restrict__ qn, __bf16* __restrict__ kn, __bf16* __restrict__ vb) {
  extern __shared__ char smem[];
  __bf16* As = (__bf16*)smem;                     // 64 x 384
  __bf16* Ws = (__bf16*)(smem + 64 * DIMC * 2);   // 32 x 384
  const int tid = threadIdx.x;
  const int rbase = blockIdx.x * 64;
  const int tile = blockIdx.y;            // 0..35
  const int sec = tile / NHEADS;          // 0=q, 1=k, 2=v
  const int hh  = tile % NHEADS;
  const int cbase = tile * 32;

#if HAVE_ASYNC_LDS
  {
    const char* wsrc = (const char*)(wq + (size_t)cbase * DIMC);
    char* wdst = (char*)Ws;
    for (int e = tid; e < (32 * DIMC * 2) / 16; e += 128)   // 1536 x 16B
      async_cp16(wsrc + e * 16, wdst + e * 16);
  }
#else
  {
    const unsigned* wsrc = (const unsigned*)(wq + (size_t)cbase * DIMC);
    unsigned* wdst = (unsigned*)Ws;
    for (int e = tid; e < 32 * DIMC / 2; e += 128) wdst[e] = wsrc[e];
  }
#endif
  for (int e = tid; e < 64 * DIMC; e += 128)
    As[e] = f2bf(x[(size_t)(rbase + e / DIMC) * DIMC + (e % DIMC)]);
  async_wait();
  __syncthreads();

  const int wid = tid >> 5, lane = tid & 31;
  const int l16 = lane & 15, half8 = (lane >> 4) << 3;
  v8f c0 = {}; v8f c1 = {};
#pragma unroll
  for (int kt = 0; kt < DIMC / 32; ++kt) {
    v16bf a  = ld_frag(As + wid * 16 * DIMC + kt * 32, DIMC);
    v16bf b0 = ld_frag(Ws + kt * 32, DIMC);
    v16bf b1 = ld_frag(Ws + 16 * DIMC + kt * 32, DIMC);
    c0 = wmma_bf16(a, b0, c0);
    c1 = wmma_bf16(a, b1, c1);
  }
  float bia0 = 0.f, bia1 = 0.f;
  if (sec == 0) { bia0 = q_bias[hh * 32 + l16]; bia1 = q_bias[hh * 32 + 16 + l16]; }
  else if (sec == 2) { bia0 = v_bias[hh * 32 + l16]; bia1 = v_bias[hh * 32 + 16 + l16]; }
#pragma unroll
  for (int i = 0; i < 8; ++i) { c0[i] += bia0; c1[i] += bia1; }
  if (sec < 2) {  // cosine-attention row normalization over the head's 32 dims
#pragma unroll
    for (int i = 0; i < 8; ++i) {
      float s = c0[i] * c0[i] + c1[i] * c1[i];
      s += __shfl_xor(s, 1, 32);
      s += __shfl_xor(s, 2, 32);
      s += __shfl_xor(s, 4, 32);
      s += __shfl_xor(s, 8, 32);
      float rn = 1.0f / fmaxf(sqrtf(s), 1e-12f);
      c0[i] *= rn; c1[i] *= rn;
    }
  }
  __bf16* dst = (sec == 0) ? qn : (sec == 1) ? kn : vb;
#pragma unroll
  for (int i = 0; i < 8; ++i) {
    int gr = rbase + wid * 16 + i + half8;
    int b = gr / NTOK, n = gr % NTOK;
    size_t o = ((size_t)(b * NHEADS + hh) * NTOK + n) * 32;
    dst[o + l16]      = f2bf(c0[i]);
    dst[o + 16 + l16] = f2bf(c1[i]);
  }
}

// ---------- K5: attention for one (b, h, 64-row block) ----------

__global__ void __launch_bounds__(256)
k_attn(const __bf16* __restrict__ qn, const __bf16* __restrict__ kn,
       const __bf16* __restrict__ vb, const float* __restrict__ bias_full,
       const float* __restrict__ mask, const float* __restrict__ logit_scale,
       float* __restrict__ attn_out) {
  extern __shared__ char smem[];
  __bf16* Ks = (__bf16*)smem;                              // 352 x 32 token-major
  __bf16* Vt = (__bf16*)(smem + NPAD * 32 * 2);            // 32 x 352 dim-major
  float*  Ss = (float*)(smem + 2 * NPAD * 32 * 2);         // 64 x 352 fp32
  __bf16* Ps = (__bf16*)(smem + 2 * NPAD * 32 * 2 + 64 * NPAD * 4);  // 64 x 352
  float*  red = (float*)(smem + 2 * NPAD * 32 * 2 + 64 * NPAD * 4 + 64 * NPAD * 2); // 256 f32

  const int rb = blockIdx.x;   // 0..5
  const int h  = blockIdx.y;
  const int b  = blockIdx.z;
  const int w  = b & 7;        // window index: B reshaped (16, 8)
  const int tid = threadIdx.x;
  const size_t bh = (size_t)(b * NHEADS + h) * NTOK * 32;

  // stage K (token-major, zero-padded to 352 tokens)
#if HAVE_ASYNC_LDS
  {
    const char* ksrc = (const char*)(kn + bh);
    char* kdst = (char*)Ks;
    for (int i = tid; i < (NTOK * 32 * 2) / 16; i += 256)   // 1372 x 16B
      async_cp16(ksrc + i * 16, kdst + i * 16);
    unsigned* kz = (unsigned*)Ks;
    for (int i = NTOK * 16 + tid; i < NPAD * 16; i += 256) kz[i] = 0u;
  }
#else
  {
    const unsigned* ksrc = (const unsigned*)(kn + bh);
    unsigned* kdst = (unsigned*)Ks;
    for (int i = tid; i < NPAD * 16; i += 256)
      kdst[i] = (i < NTOK * 16) ? ksrc[i] : 0u;
  }
#endif
  // stage V transposed to dim-major
  {
    const __bf16* vsrc = vb + bh;
    for (int e = tid; e < 32 * NPAD; e += 256) {
      int d = e / NPAD, t = e % NPAD;
      Vt[e] = (t < NTOK) ? vsrc[t * 32 + d] : f2bf(0.f);
    }
  }
  async_wait();
  __syncthreads();

  const int wid = tid >> 5, lane = tid & 31;
  const int mt = wid >> 1, par = wid & 1;
  const int l16 = lane & 15, half8 = (lane >> 4) << 3;
  const float scale = __expf(fminf(logit_scale[h], 4.605170186f));  // ln(100)

  // A fragment: 16 query rows (K = d = 32, one WMMA K-step), OOB rows -> 0
  v16bf a;
  {
    int token = rb * 64 + mt * 16 + l16;
    if (token < NTOK) {
      const __bf16* p = qn + bh + (size_t)token * 32 + ((lane >> 4) << 3);
#pragma unroll
      for (int i = 0; i < 8; ++i) { a[i] = p[i]; a[i + 8] = p[16 + i]; }
    } else {
#pragma unroll
      for (int i = 0; i < 16; ++i) a[i] = f2bf(0.f);
    }
  }

  const float* bsrc = bias_full + (size_t)h * NN;
  const float* msrc = mask + (size_t)w * NN;
  for (int nt = par; nt < NPAD / 16; nt += 2) {
    v16bf bfr = ld_frag(Ks + nt * 16 * 32, 32);
    v8f c = {};
    c = wmma_bf16(a, bfr, c);
    int n = nt * 16 + l16;
#pragma unroll
    for (int i = 0; i < 8; ++i) {
      int lm = mt * 16 + i + half8;
      int m = rb * 64 + lm;
      float add = (m < NTOK && n < NTOK)
                      ? bsrc[(size_t)m * NTOK + n] + msrc[(size_t)m * NTOK + n]
                      : -1e30f;
      Ss[lm * NPAD + n] = c[i] * scale + add;
    }
  }
  __syncthreads();

  // softmax: 4 threads per row, 88 columns each (4*88 = 352)
  {
    const int r = tid & 63, qq = tid >> 6;
    const int cb = qq * 88;
    float* row = Ss + r * NPAD;
    float mx = -1e30f;
    for (int c = cb; c < cb + 88; ++c)
      if (c < NTOK) mx = fmaxf(mx, row[c]);
    red[tid] = mx;
    __syncthreads();
    mx = fmaxf(fmaxf(red[r], red[64 + r]), fmaxf(red[128 + r], red[192 + r]));
    float sum = 0.f;
    for (int c = cb; c < cb + 88; ++c) {
      float e = (c < NTOK) ? __expf(row[c] - mx) : 0.f;
      row[c] = e; sum += e;
    }
    __syncthreads();
    red[tid] = sum;
    __syncthreads();
    float inv = 1.0f / (red[r] + red[64 + r] + red[128 + r] + red[192 + r]);
    __bf16* prow = Ps + r * NPAD;
    for (int c = cb; c < cb + 88; ++c) prow[c] = f2bf(row[c] * inv);
  }
  __syncthreads();

  // O = P @ V : wave owns (mt, d-half), 11 K-steps over 352 padded tokens
  const int nt2 = wid & 1;
  v8f o = {};
#pragma unroll
  for (int kt = 0; kt < NPAD / 32; ++kt) {
    v16bf pa  = ld_frag(Ps + mt * 16 * NPAD + kt * 32, NPAD);
    v16bf vfr = ld_frag(Vt + nt2 * 16 * NPAD + kt * 32, NPAD);
    o = wmma_bf16(pa, vfr, o);
  }
#pragma unroll
  for (int i = 0; i < 8; ++i) {
    int token = rb * 64 + mt * 16 + i + half8;
    if (token < NTOK) {
      int dcol = h * 32 + nt2 * 16 + l16;
      attn_out[(size_t)(b * NTOK + token) * DIMC + dcol] = o[i];
    }
  }
}

// ---------- K6: output projection ----------

__global__ void __launch_bounds__(128)
k_proj_gemm(const float* __restrict__ ain, const __bf16* __restrict__ wp,
            const float* __restrict__ proj_b, float* __restrict__ out) {
  extern __shared__ char smem[];
  __bf16* As = (__bf16*)smem;
  __bf16* Ws = (__bf16*)(smem + 64 * DIMC * 2);
  const int tid = threadIdx.x;
  const int rbase = blockIdx.x * 64;
  const int cbase = blockIdx.y * 32;

#if HAVE_ASYNC_LDS
  {
    const char* wsrc = (const char*)(wp + (size_t)cbase * DIMC);
    char* wdst = (char*)Ws;
    for (int e = tid; e < (32 * DIMC * 2) / 16; e += 128)
      async_cp16(wsrc + e * 16, wdst + e * 16);
  }
#else
  {
    const unsigned* wsrc = (const unsigned*)(wp + (size_t)cbase * DIMC);
    unsigned* wdst = (unsigned*)Ws;
    for (int e = tid; e < 32 * DIMC / 2; e += 128) wdst[e] = wsrc[e];
  }
#endif
  for (int e = tid; e < 64 * DIMC; e += 128)
    As[e] = f2bf(ain[(size_t)(rbase + e / DIMC) * DIMC + (e % DIMC)]);
  async_wait();
  __syncthreads();

  const int wid = tid >> 5, lane = tid & 31;
  const int l16 = lane & 15, half8 = (lane >> 4) << 3;
  v8f c0 = {}; v8f c1 = {};
#pragma unroll
  for (int kt = 0; kt < DIMC / 32; ++kt) {
    v16bf a  = ld_frag(As + wid * 16 * DIMC + kt * 32, DIMC);
    v16bf b0 = ld_frag(Ws + kt * 32, DIMC);
    v16bf b1 = ld_frag(Ws + 16 * DIMC + kt * 32, DIMC);
    c0 = wmma_bf16(a, b0, c0);
    c1 = wmma_bf16(a, b1, c1);
  }
  float pb0 = proj_b[cbase + l16], pb1 = proj_b[cbase + 16 + l16];
#pragma unroll
  for (int i = 0; i < 8; ++i) {
    int gr = rbase + wid * 16 + i + half8;
    out[(size_t)gr * DIMC + cbase + l16]      = c0[i] + pb0;
    out[(size_t)gr * DIMC + cbase + 16 + l16] = c1[i] + pb1;
  }
}

// ---------- launch ----------

extern "C" void kernel_launch(void* const* d_in, const int* in_sizes, int n_in,
                              void* d_out, int out_size, void* d_ws, size_t ws_size,
                              hipStream_t stream) {
  (void)in_sizes; (void)n_in; (void)out_size; (void)ws_size;
  const float* x    = (const float*)d_in[0];
  const float* mask = (const float*)d_in[1];
  const float* qkvw = (const float*)d_in[2];
  const float* qb   = (const float*)d_in[3];
  const float* vbi  = (const float*)d_in[4];
  const float* ls   = (const float*)d_in[5];
  const float* w1   = (const float*)d_in[6];
  const float* b1   = (const float*)d_in[7];
  const float* w2   = (const float*)d_in[8];
  const float* pw   = (const float*)d_in[9];
  const float* pb   = (const float*)d_in[10];
  const float* tbl  = (const float*)d_in[11];
  const int*   rpi  = (const int*)d_in[12];
  float* out = (float*)d_out;

  char* ws = (char*)d_ws;
  size_t off = 0;
  auto alloc = [&](size_t bytes) {
    char* p = ws + off;
    off = (off + bytes + 255) & ~(size_t)255;
    return p;
  };
  __bf16* wq_bf = (__bf16*)alloc((size_t)3 * DIMC * DIMC * 2);
  __bf16* wp_bf = (__bf16*)alloc((size_t)DIMC * DIMC * 2);
  float*  tcpb  = (float*)alloc((size_t)TBL * NHEADS * 4);
  float*  biasf = (float*)alloc((size_t)NHEADS * NN * 4);
  __bf16* qn    = (__bf16*)alloc((size_t)BWIN * NHEADS * NTOK * 32 * 2);
  __bf16* kn    = (__bf16*)alloc((size_t)BWIN * NHEADS * NTOK * 32 * 2);
  __bf16* vb    = (__bf16*)alloc((size_t)BWIN * NHEADS * NTOK * 32 * 2);
  float*  aout  = (float*)alloc((size_t)ROWS * DIMC * 4);

  const size_t lds_gemm = (size_t)64 * DIMC * 2 + (size_t)32 * DIMC * 2;       // 73728 B
  const size_t lds_attn = (size_t)2 * NPAD * 32 * 2 + (size_t)64 * NPAD * 4
                        + (size_t)64 * NPAD * 2 + 256 * 4;                     // 181248 B

  k_convert<<<dim3((3 * DIMC * DIMC + 255) / 256), dim3(256), 0, stream>>>(
      qkvw, pw, wq_bf, wp_bf);
  k_cpb<<<dim3((TBL * NHEADS + 255) / 256), dim3(256), 0, stream>>>(
      tbl, w1, b1, w2, tcpb);
  k_bias<<<dim3((NHEADS * NN + 255) / 256), dim3(256), 0, stream>>>(
      tcpb, rpi, biasf);
  k_qkv_gemm<<<dim3(ROWS / 64, 36), dim3(128), lds_gemm, stream>>>(
      x, wq_bf, qb, vbi, qn, kn, vb);
  k_attn<<<dim3(6, NHEADS, BWIN), dim3(256), lds_attn, stream>>>(
      qn, kn, vb, biasf, mask, ls, aout);
  k_proj_gemm<<<dim3(ROWS / 64, DIMC / 32), dim3(128), lds_gemm, stream>>>(
      aout, wp_bf, pb, out);
}
